// GraphKAN_47828755808716
// MI455X (gfx1250) — matrix operand order repaired and checked by reading, not compiled, not run on hardware
//
#include <hip/hip_runtime.h>
#include <hip/hip_bf16.h>

typedef _Float16 f16;
typedef _Float16 v16h __attribute__((ext_vector_type(16)));
typedef _Float16 v8h  __attribute__((ext_vector_type(8)));
typedef float    v8f  __attribute__((ext_vector_type(8)));

#define N_NODES 100000
#define N_EDGES 1600000
#define F_IN    64
#define F_OUT   64
#define N_BASES 8
#define KDIM    (F_IN * N_BASES)   // 512
#define LN_EPS  1e-5f
// phi row stride in LDS (halves); +8 halves pad to spread banks
#define LDS_STRIDE (KDIM + 8)

// ---------------------------------------------------------------- zero scratch
__global__ void gk_zero_f32(float* __restrict__ p, long n) {
    long i = (long)blockIdx.x * blockDim.x + threadIdx.x;
    long stride = (long)gridDim.x * blockDim.x;
    for (; i < n; i += stride) p[i] = 0.0f;
}

// ---------------------------------------------------------------- degree accum
// deg[s] = sum of edge weights with src==s, plus 1.0 for the self loop
__global__ void gk_degree(const long long* __restrict__ ei,
                          const float* __restrict__ ew,
                          float* __restrict__ deg) {
    long i = (long)blockIdx.x * blockDim.x + threadIdx.x;
    long total = (long)N_EDGES + N_NODES;
    long stride = (long)gridDim.x * blockDim.x;
    for (; i < total; i += stride) {
        if (i < N_EDGES) {
            atomicAdd(&deg[(int)ei[i]], ew[i]);           // src = ei[0][i]
        } else {
            atomicAdd(&deg[(int)(i - N_EDGES)], 1.0f);    // self loop
        }
    }
}

// ---------------------------------------------------------------- aggregation
// One wave (32 lanes) per edge; lane handles features {lane, lane+32}.
__global__ __launch_bounds__(256)
void gk_aggregate(const long long* __restrict__ ei,
                  const float* __restrict__ ew,
                  const float* __restrict__ x,
                  const float* __restrict__ deg,
                  float* __restrict__ agg) {
    long tid  = (long)blockIdx.x * blockDim.x + threadIdx.x;
    long wid  = tid >> 5;
    int  lane = (int)(tid & 31);
    long total = (long)N_EDGES + N_NODES;
    if (wid >= total) return;
    int s, d; float w;
    if (wid < N_EDGES) {
        s = (int)ei[wid];
        d = (int)ei[N_EDGES + wid];
        w = ew[wid];
    } else {
        s = (int)(wid - N_EDGES); d = s; w = 1.0f;        // self loop
    }
    float dv = deg[s];
    dv = dv < 1.0f ? 1.0f : dv;
    float wn = w / dv;
    const float* xr = x + (size_t)d * F_IN;
    float* ar = agg + (size_t)s * F_IN;
    atomicAdd(&ar[lane],      wn * xr[lane]);
    atomicAdd(&ar[lane + 32], wn * xr[lane + 32]);
}

// ---------------------------------------------------------------- W -> f16 B-fragments
// Bpack[kt][nt][lane][j] : lane<16 -> N=nt*16+lane, K=kt*32+j ; lane>=16 -> K=kt*32+16+j
__global__ void gk_pack_w(const float* __restrict__ W, f16* __restrict__ Bp) {
    int t = blockIdx.x * blockDim.x + threadIdx.x;        // 16*4*32 = 2048 threads
    if (t >= 16 * 4 * 32) return;
    int lane = t & 31;
    int nt   = (t >> 5) & 3;
    int kt   = t >> 7;
    int n    = nt * 16 + (lane & 15);
    int kb   = kt * 32 + ((lane < 16) ? 0 : 16);
    f16* dst = Bp + (size_t)t * 16;
    #pragma unroll
    for (int j = 0; j < 16; ++j)
        dst[j] = (f16)W[(size_t)(kb + j) * F_OUT + n];
}

// ---------------------------------------------------------------- fused LN + RBF + WMMA GEMM
// Block = 128 threads (4 waves) handles 16 nodes.
// Phase 1: 8 lanes/node compute LayerNorm + Gaussian RBF -> phi in LDS (f16, 16x512).
// Phase 2: wave nt computes the 16x16 output tile for columns nt*16..+15 with
//          16 x v_wmma_f32_16x16x32_f16 over K=512.
__global__ __launch_bounds__(128)
void gk_fused(const float* __restrict__ agg,
              const float* __restrict__ gamma,
              const float* __restrict__ beta,
              const f16*  __restrict__ Bp,
              const float* __restrict__ bias,
              float* __restrict__ out) {
    __shared__ f16 phi[16 * LDS_STRIDE];   // 16.6 KB

    const int tid   = threadIdx.x;
    const int node_in_tile = tid >> 3;     // 0..15
    const int fg    = tid & 7;             // feature group: 8 features each
    const int node0 = blockIdx.x * 16;
    const int node  = node0 + node_in_tile;

    // ---- LayerNorm over 64 features (8 lanes x 8 features) ----
    float a[8];
    const float* arow = agg + (size_t)node * F_IN + fg * 8;
    #pragma unroll
    for (int j = 0; j < 8; ++j) a[j] = arow[j];

    float s = 0.0f;
    #pragma unroll
    for (int j = 0; j < 8; ++j) s += a[j];
    s += __shfl_xor(s, 1); s += __shfl_xor(s, 2); s += __shfl_xor(s, 4);
    float mu = s * (1.0f / 64.0f);

    float v = 0.0f;
    #pragma unroll
    for (int j = 0; j < 8; ++j) { float dlt = a[j] - mu; v += dlt * dlt; }
    v += __shfl_xor(v, 1); v += __shfl_xor(v, 2); v += __shfl_xor(v, 4);
    float rstd = rsqrtf(v * (1.0f / 64.0f) + LN_EPS);

    // ---- RBF expansion: centers linspace(-1,1,8), 1/(2*sigma^2) = 49/2 ----
    #pragma unroll
    for (int j = 0; j < 8; ++j) {
        int f = fg * 8 + j;
        float h = (a[j] - mu) * rstd * gamma[f] + beta[f];
        v8h p;
        #pragma unroll
        for (int b = 0; b < 8; ++b) {
            float c  = -1.0f + (2.0f / 7.0f) * (float)b;
            float dd = h - c;
            p[b] = (f16)__expf(-24.5f * dd * dd);
        }
        *(v8h*)&phi[node_in_tile * LDS_STRIDE + f * 8] = p;
    }
    __syncthreads();

    // ---- WMMA phase: wave = output column tile ----
    const int wave = tid >> 5;             // nt: 0..3
    const int lane = tid & 31;
    const int row  = lane & 15;
    const int koff = (lane < 16) ? 0 : 16; // contiguous K: k = 16h + s

    v8f acc = {0.f, 0.f, 0.f, 0.f, 0.f, 0.f, 0.f, 0.f};
    #pragma unroll
    for (int kt = 0; kt < 16; ++kt) {
        int kbase = kt * 32 + koff;
        v8h lo = *(const v8h*)&phi[row * LDS_STRIDE + kbase];        // K {0..7}
        v8h hi = *(const v8h*)&phi[row * LDS_STRIDE + kbase + 8];    // K {8..15}
        v16h afrag = __builtin_shufflevector(lo, hi,
            0, 1, 2, 3, 4, 5, 6, 7, 8, 9, 10, 11, 12, 13, 14, 15);
        v16h bfrag = *(const v16h*)(Bp + ((size_t)(kt * 4 + wave) * 32 + lane) * 16);
        acc = __builtin_amdgcn_wmma_f32_16x16x32_f16(
            false, afrag, false, bfrag, (short)0, acc, false, false);
    }

    // ---- store (C layout: lanes 0-15 -> rows 0..7, lanes 16-31 -> rows 8..15) ----
    int col   = wave * 16 + (lane & 15);
    int rbase = (lane < 16) ? 0 : 8;
    float bv  = bias[col];
    #pragma unroll
    for (int r = 0; r < 8; ++r)
        out[(size_t)(node0 + rbase + r) * F_OUT + col] = acc[r] + bv;
}

// ---------------------------------------------------------------- launcher
extern "C" void kernel_launch(void* const* d_in, const int* in_sizes, int n_in,
                              void* d_out, int out_size, void* d_ws, size_t ws_size,
                              hipStream_t stream) {
    const float*     x     = (const float*)d_in[0];
    const long long* ei    = (const long long*)d_in[1];   // int64 (2, E)
    const float*     ew    = (const float*)d_in[2];
    const float*     gamma = (const float*)d_in[3];
    const float*     beta  = (const float*)d_in[4];
    const float*     W     = (const float*)d_in[5];       // (64, 8, 64) -> (512, 64)
    const float*     bias  = (const float*)d_in[6];
    float*           out   = (float*)d_out;

    // workspace layout (all offsets 32B aligned): deg | agg | Bpack
    float* wsf = (float*)d_ws;
    float* deg = wsf;                                     // N floats      (400000 B)
    float* agg = wsf + N_NODES;                           // N*64 floats   (25.6 MB)
    f16*   Bp  = (f16*)(wsf + N_NODES + (size_t)N_NODES * F_IN); // 64 KB

    // 1) zero deg + agg (contiguous)
    long nzero = (long)N_NODES + (long)N_NODES * F_IN;    // 6.5M floats
    gk_zero_f32<<<4096, 256, 0, stream>>>(wsf, nzero);

    // 2) pack weights into per-lane f16 B fragments
    gk_pack_w<<<8, 256, 0, stream>>>(W, Bp);

    // 3) degrees
    long total_e = (long)N_EDGES + N_NODES;
    int degBlocks = (int)((total_e + 255) / 256);
    gk_degree<<<degBlocks, 256, 0, stream>>>(ei, ew, deg);

    // 4) edge aggregation: one wave per edge
    long athreads = total_e * 32;
    int aggBlocks = (int)((athreads + 255) / 256);
    gk_aggregate<<<aggBlocks, 256, 0, stream>>>(ei, ew, x, deg, agg);

    // 5) fused LayerNorm + RBF + WMMA contraction (N_NODES % 16 == 0 -> 6250 tiles)
    gk_fused<<<N_NODES / 16, 128, 0, stream>>>(agg, gamma, beta, Bp, bias, out);
}
